// PyGT_TGCN_62835371541091
// MI455X (gfx1250) — matrix operand order repaired and checked by reading, not compiled, not run on hardware
//
#include <hip/hip_runtime.h>
#include <hip/hip_bf16.h>

// ---------------------------------------------------------------------------
// CDNA5 / gfx1250 TGCN implementation.
// All GEMMs use v_wmma_f32_16x16x32_bf16 (wave32, f32 accumulate).
// C-fragments are staged through LDS so global stores are coalesced b128.
// ---------------------------------------------------------------------------

typedef __attribute__((ext_vector_type(16))) __bf16 v16bf;
typedef __attribute__((ext_vector_type(4)))  __bf16 v4bf;
typedef __attribute__((ext_vector_type(8)))  float  v8f;

#define WMMA_BF16(a, b, c) \
  __builtin_amdgcn_wmma_f32_16x16x32_bf16(false, (a), false, (b), (short)0, (c), false, false)

// A-fragment (16x32 bf16, M x K) built from a row-major f32 tile in LDS.
// Layout (ISA 7.12.2): lanes 0-15 hold row M=lane, K in {kbase+0..7, kbase+16..23};
// lanes 16-31 hold row M=lane-16, K in {kbase+8..15, kbase+24..31}.
__device__ __forceinline__ v16bf frag_a_lds(const float* p0, int ld, int lane, int kbase) {
  const float* p = p0 + (lane & 15) * ld + kbase + ((lane & 16) ? 8 : 0);
  v16bf f;
#pragma unroll
  for (int j = 0; j < 8; ++j) {
    f[j]     = (__bf16)p[j];
    f[j + 8] = (__bf16)p[j + 16];
  }
  return f;
}

// B-fragment from pre-swizzled weights: frag[((kk*nTiles+n)*32+lane)*16 + j]
// where element j of lane l is W[kk*32 + ((l>>4)&1)*8 + (j>>3)*16 + (j&7)][n*16 + (l&15)].
__device__ __forceinline__ v16bf frag_b_glb(const __bf16* fragBase, int nTiles,
                                            int kk, int n, int lane) {
  return *(const v16bf*)(fragBase + (((size_t)(kk * nTiles + n)) * 32 + lane) * 16);
}

__device__ __forceinline__ float sigmoidf_(float x) {
  return 1.0f / (1.0f + __expf(-x));
}

// ---------------------------------------------------------------------------
// Degree / norm kernels.  deg buffer doubles as dinv buffer.
// ---------------------------------------------------------------------------
__global__ void k_deg_init(float* __restrict__ deg, int N) {
  int i = blockIdx.x * blockDim.x + threadIdx.x;
  if (i < N) deg[i] = 1.0f;  // self-loop weight
}

__global__ void k_deg_atomic(const long long* __restrict__ ei,
                             const float* __restrict__ ew,
                             float* __restrict__ deg, int E) {
  int e = blockIdx.x * blockDim.x + threadIdx.x;
  if (e < E) atomicAdd(&deg[(int)ei[(size_t)E + e]], ew[e]);
}

__global__ void k_dinv(float* __restrict__ dinv, int N) {
  int i = blockIdx.x * blockDim.x + threadIdx.x;
  if (i < N) {
    float d = dinv[i];
    dinv[i] = (d > 0.0f) ? rsqrtf(d) : 0.0f;
  }
}

// ---------------------------------------------------------------------------
// Weight swizzle into WMMA B-fragment order (f32 -> bf16).
// ---------------------------------------------------------------------------
__global__ void k_prep(const float* __restrict__ W, __bf16* __restrict__ frag,
                       int K, int Ncols) {
  int idx = blockIdx.x * blockDim.x + threadIdx.x;
  if (idx >= K * Ncols) return;
  int j    = idx & 15;
  int lane = (idx >> 4) & 31;
  int rest = idx >> 9;
  int nT   = Ncols >> 4;
  int n    = rest % nT;
  int kk   = rest / nT;
  int Kidx = kk * 32 + ((lane >> 4) & 1) * 8 + (j >> 3) * 16 + (j & 7);
  int col  = n * 16 + (lane & 15);
  frag[idx] = (__bf16)W[Kidx * Ncols + col];
}

// ---------------------------------------------------------------------------
// GEMM1: h_g = bf16(X @ W_g), g in {z,r,h}.  X is N x 64, W is 64 x 128.
// One wave per 16-row tile; 4 waves per block.  Output staged via LDS so the
// global stores are coalesced b128 on a wave-uniform fast path.
// ---------------------------------------------------------------------------
__global__ void __launch_bounds__(128) k_gemm1(const float* __restrict__ x,
                                               const __bf16* __restrict__ fragW1,
                                               __bf16* __restrict__ hB, int N) {
  __shared__ __align__(16) float  sx[4][16 * 64];   // 16 KB
  __shared__ __align__(16) __bf16 so[4][16 * 128];  // 16 KB
  const int lane  = threadIdx.x & 31;
  const int w     = threadIdx.x >> 5;
  const int tiles = (N + 15) >> 4;
  int tile = blockIdx.x * 4 + w;
  const bool valid = tile < tiles;
  if (tile >= tiles) tile = tiles - 1;
  const int row0 = tile << 4;
  const bool full = valid && (row0 + 16 <= N);

  // stage X tile: 1024 f32 = 256 float4
  for (int t = lane; t < 256; t += 32) {
    int r = row0 + (t >> 4);
    if (r >= N) r = N - 1;
    *(float4*)(sx[w] + t * 4) = *(const float4*)(x + (size_t)r * 64 + (t & 15) * 4);
  }
  __syncthreads();

  const int colL = lane & 15;
  const int rL   = (lane & 16) ? 8 : 0;

  for (int g = 0; g < 3; ++g) {
    const __bf16* fw = fragW1 + (size_t)g * (2 * 8 * 32 * 16);
    for (int n = 0; n < 8; ++n) {
      v8f acc = {};
#pragma unroll
      for (int kk = 0; kk < 2; ++kk) {
        v16bf a = frag_a_lds(sx[w], 64, lane, kk * 32);
        v16bf b = frag_b_glb(fw, 8, kk, n, lane);
        acc = WMMA_BF16(a, b, acc);
      }
      int col = n * 16 + colL;
#pragma unroll
      for (int i = 0; i < 8; ++i) so[w][(rL + i) * 128 + col] = (__bf16)acc[i];
    }
    __syncthreads();
    // coalesced store: 2048 bf16 = 4 KB = 256 uint4
    __bf16* gbase = hB + (size_t)g * N * 128 + (size_t)row0 * 128;
    if (full) {
      uint4* gp = (uint4*)gbase;
      const uint4* ls = (const uint4*)so[w];
      for (int t = lane; t < 256; t += 32) gp[t] = ls[t];
    } else if (valid) {
      for (int idx = lane; idx < 2048; idx += 32) {
        int r = row0 + (idx >> 7);
        if (r < N) hB[(size_t)g * N * 128 + (size_t)r * 128 + (idx & 127)] = so[w][idx];
      }
    }
    __syncthreads();
  }
}

// ---------------------------------------------------------------------------
// agg_g[i,c] = dinv[i]^2 * h_g[i,c]   (self-loop term; also initializes agg)
// Vectorized x4.
// ---------------------------------------------------------------------------
__global__ void k_agg_init(const float* __restrict__ dinv,
                           const __bf16* __restrict__ hB,
                           float* __restrict__ agg, int N) {
  size_t q     = (size_t)blockIdx.x * blockDim.x + threadIdx.x;
  size_t total = (size_t)3 * N * 32;  // quads
  if (q >= total) return;
  size_t idx  = q * 4;
  size_t node = (idx >> 7) % (size_t)N;
  float d  = dinv[node];
  float dd = d * d;
  v4bf h = *(const v4bf*)(hB + idx);
  float4 o = {dd * (float)h[0], dd * (float)h[1], dd * (float)h[2], dd * (float)h[3]};
  *(float4*)(agg + idx) = o;
}

// ---------------------------------------------------------------------------
// Edge scatter: one wave per 4 edges, lane owns 4 of 128 features, 3 gates.
// agg_g[dst,:] += dinv[src]*ew*dinv[dst] * h_g[src,:]
// Prefetch upcoming source rows (global_prefetch_b8) to hide L2 latency on
// the random gather — this kernel is the bandwidth bottleneck.
// ---------------------------------------------------------------------------
#define EPW 4
__global__ void __launch_bounds__(256) k_scatter(const long long* __restrict__ ei,
                                                 const float* __restrict__ ew,
                                                 const float* __restrict__ dinv,
                                                 const __bf16* __restrict__ hB,
                                                 float* __restrict__ agg,
                                                 int E, int N) {
  long long wid = ((long long)blockIdx.x * blockDim.x + threadIdx.x) >> 5;
  int lane = threadIdx.x & 31;
  long long e0 = wid * EPW;
  if (e0 >= E) return;
  int ne = (E - e0 < EPW) ? (int)(E - e0) : EPW;
  int c = lane * 4;

  for (int t = 0; t < ne; ++t) {
    int s = (int)ei[e0 + t];
    const __bf16* hp = hB + (size_t)s * 128 + c;
    __builtin_prefetch(hp, 0, 3);
    __builtin_prefetch(hp + (size_t)N * 128, 0, 3);
    __builtin_prefetch(hp + (size_t)2 * N * 128, 0, 3);
  }

  for (int t = 0; t < ne; ++t) {
    long long e = e0 + t;
    int src = (int)ei[e];
    int dst = (int)ei[(size_t)E + e];
    float norm = dinv[src] * ew[e] * dinv[dst];
#pragma unroll
    for (int g = 0; g < 3; ++g) {
      v4bf hv = *(const v4bf*)(hB + ((size_t)g * N + src) * 128 + c);
      float* ap = agg + ((size_t)g * N + dst) * 128 + c;
      atomicAdd(ap + 0, norm * (float)hv[0]);
      atomicAdd(ap + 1, norm * (float)hv[1]);
      atomicAdd(ap + 2, norm * (float)hv[2]);
      atomicAdd(ap + 3, norm * (float)hv[3]);
    }
  }
}

// ---------------------------------------------------------------------------
// Fused gate + head kernel.  One wave owns a 16-row tile.
//   Z  = sigmoid([az+bz, H]  @ Wlz + blz)
//   R  = sigmoid([ar+br, H]  @ Wlr + blr)
//   Ht = tanh   ([ah+bh, H*R]@ Wlh + blh)
//   Hn = Z*H + (1-Z)*Ht ; y = relu(Hn) @ Whead + bhead
// 208 WMMAs per wave.  48 KB static LDS per 2-wave block.
// All global stores staged through LDS -> coalesced b128.
// ---------------------------------------------------------------------------
__global__ void __launch_bounds__(64) k_gates(
    const float* __restrict__ agg, const float* __restrict__ Hin,
    const float* __restrict__ bz, const float* __restrict__ br,
    const float* __restrict__ bh, const __bf16* __restrict__ fragWl,
    const float* __restrict__ blz, const float* __restrict__ blr,
    const float* __restrict__ blh, const __bf16* __restrict__ fragHead,
    const float* __restrict__ bhead, float* __restrict__ outY,
    float* __restrict__ outH, int N) {
  __shared__ __align__(16) float smem[2 * 3 * 2048];
  const int lane  = threadIdx.x & 31;
  const int w     = threadIdx.x >> 5;
  const int tiles = (N + 15) >> 4;
  int tile = blockIdx.x * 2 + w;
  const bool valid = tile < tiles;
  if (tile >= tiles) tile = tiles - 1;
  const int row0 = tile << 4;
  const bool full = valid && (row0 + 16 <= N);

  float* bufA = smem + w * 3 * 2048;  // GCN part of concat A (per gate)
  float* bufH = bufA + 2048;          // H tile
  float* bufC = bufH + 2048;          // H*R tile, then H_new / y staging

  // stage H and bufA(z): 2048 f32 = 512 float4 each
  for (int t = lane; t < 512; t += 32) {
    int r = row0 + (t >> 5);
    if (r >= N) r = N - 1;
    int cq = (t & 31) * 4;
    *(float4*)(bufH + t * 4) = *(const float4*)(Hin + (size_t)r * 128 + cq);
    float4 av  = *(const float4*)(agg + (size_t)r * 128 + cq);
    float4 bv  = *(const float4*)(bz + cq);
    av.x += bv.x; av.y += bv.y; av.z += bv.z; av.w += bv.w;
    *(float4*)(bufA + t * 4) = av;
  }
  __syncthreads();

  const int colL = lane & 15;
  const int rL   = (lane & 16) ? 8 : 0;

  // ---- Z ----
  v8f zacc[8];
#pragma unroll
  for (int n = 0; n < 8; ++n) {
    v8f acc = {};
#pragma unroll
    for (int kk = 0; kk < 8; ++kk) {
      const float* ab = (kk < 4) ? bufA : bufH;
      v16bf a = frag_a_lds(ab, 128, lane, (kk & 3) * 32);
      v16bf b = frag_b_glb(fragWl, 8, kk, n, lane);
      acc = WMMA_BF16(a, b, acc);
    }
    float bias = blz[n * 16 + colL];
#pragma unroll
    for (int i = 0; i < 8; ++i) zacc[n][i] = sigmoidf_(acc[i] + bias);
  }
  __syncthreads();

  // ---- R, write H*R into bufC ----
  for (int t = lane; t < 512; t += 32) {
    int r = row0 + (t >> 5);
    if (r >= N) r = N - 1;
    int cq = (t & 31) * 4;
    float4 av = *(const float4*)(agg + (size_t)N * 128 + (size_t)r * 128 + cq);
    float4 bv = *(const float4*)(br + cq);
    av.x += bv.x; av.y += bv.y; av.z += bv.z; av.w += bv.w;
    *(float4*)(bufA + t * 4) = av;
  }
  __syncthreads();

  const __bf16* fr = fragWl + (size_t)(8 * 8 * 32 * 16);
  for (int n = 0; n < 8; ++n) {
    v8f acc = {};
#pragma unroll
    for (int kk = 0; kk < 8; ++kk) {
      const float* ab = (kk < 4) ? bufA : bufH;
      v16bf a = frag_a_lds(ab, 128, lane, (kk & 3) * 32);
      v16bf b = frag_b_glb(fr, 8, kk, n, lane);
      acc = WMMA_BF16(a, b, acc);
    }
    float bias = blr[n * 16 + colL];
    int col = n * 16 + colL;
#pragma unroll
    for (int i = 0; i < 8; ++i) {
      float rr = sigmoidf_(acc[i] + bias);
      bufC[(rL + i) * 128 + col] = rr * bufH[(rL + i) * 128 + col];
    }
  }
  __syncthreads();

  // ---- Ht, combine into H_new (kept in zacc) ----
  for (int t = lane; t < 512; t += 32) {
    int r = row0 + (t >> 5);
    if (r >= N) r = N - 1;
    int cq = (t & 31) * 4;
    float4 av = *(const float4*)(agg + (size_t)2 * N * 128 + (size_t)r * 128 + cq);
    float4 bv = *(const float4*)(bh + cq);
    av.x += bv.x; av.y += bv.y; av.z += bv.z; av.w += bv.w;
    *(float4*)(bufA + t * 4) = av;
  }
  __syncthreads();

  const __bf16* fh = fragWl + (size_t)(2 * 8 * 8 * 32 * 16);
#pragma unroll
  for (int n = 0; n < 8; ++n) {
    v8f acc = {};
#pragma unroll
    for (int kk = 0; kk < 8; ++kk) {
      const float* ab = (kk < 4) ? bufA : bufC;
      v16bf a = frag_a_lds(ab, 128, lane, (kk & 3) * 32);
      v16bf b = frag_b_glb(fh, 8, kk, n, lane);
      acc = WMMA_BF16(a, b, acc);
    }
    float bias = blh[n * 16 + colL];
    int col = n * 16 + colL;
#pragma unroll
    for (int i = 0; i < 8; ++i) {
      float ht = tanhf(acc[i] + bias);
      float hv = bufH[(rL + i) * 128 + col];
      float z  = zacc[n][i];
      zacc[n][i] = z * hv + (1.0f - z) * ht;  // H_new
    }
  }
  __syncthreads();

  // ---- stage H_new into bufC (for store) and relu(H_new) into bufA ----
#pragma unroll
  for (int n = 0; n < 8; ++n) {
    int col = n * 16 + colL;
#pragma unroll
    for (int i = 0; i < 8; ++i) {
      float hn = zacc[n][i];
      bufC[(rL + i) * 128 + col] = hn;
      bufA[(rL + i) * 128 + col] = fmaxf(hn, 0.0f);
    }
  }
  __syncthreads();

  // coalesced H_new store: 2048 f32 = 512 float4
  if (full) {
    float4* gp = (float4*)(outH + (size_t)row0 * 128);
    for (int t = lane; t < 512; t += 32) gp[t] = *(const float4*)(bufC + t * 4);
  } else if (valid) {
    for (int idx = lane; idx < 2048; idx += 32) {
      int r = row0 + (idx >> 7);
      if (r < N) outH[(size_t)r * 128 + (idx & 127)] = bufC[idx];
    }
  }

  // ---- head GEMM: y = relu(H_new) @ Whead + bhead ----
  v8f yacc[4];
#pragma unroll
  for (int n = 0; n < 4; ++n) {
    v8f acc = {};
#pragma unroll
    for (int kk = 0; kk < 4; ++kk) {
      v16bf a = frag_a_lds(bufA, 128, lane, kk * 32);
      v16bf b = frag_b_glb(fragHead, 4, kk, n, lane);
      acc = WMMA_BF16(a, b, acc);
    }
    float bias = bhead[n * 16 + colL];
#pragma unroll
    for (int i = 0; i < 8; ++i) yacc[n][i] = acc[i] + bias;
  }
  __syncthreads();  // bufC reads (H_new store) done before overwrite

#pragma unroll
  for (int n = 0; n < 4; ++n) {
    int col = n * 16 + colL;
#pragma unroll
    for (int i = 0; i < 8; ++i) bufC[(rL + i) * 64 + col] = yacc[n][i];
  }
  __syncthreads();

  // coalesced y store: 1024 f32 = 256 float4
  if (full) {
    float4* gp = (float4*)(outY + (size_t)row0 * 64);
    for (int t = lane; t < 256; t += 32) gp[t] = *(const float4*)(bufC + t * 4);
  } else if (valid) {
    for (int idx = lane; idx < 1024; idx += 32) {
      int r = row0 + (idx >> 6);
      if (r < N) outY[(size_t)r * 64 + (idx & 63)] = bufC[idx];
    }
  }
}

// ---------------------------------------------------------------------------
extern "C" void kernel_launch(void* const* d_in, const int* in_sizes, int n_in,
                              void* d_out, int out_size, void* d_ws, size_t ws_size,
                              hipStream_t stream) {
  (void)n_in; (void)out_size; (void)ws_size;
  const float*     node_feat = (const float*)d_in[0];
  const float*     edge_w    = (const float*)d_in[1];
  const float*     hidden    = (const float*)d_in[2];
  const long long* edge_idx  = (const long long*)d_in[3];
  const float* Wz = (const float*)d_in[4];  const float* bz = (const float*)d_in[5];
  const float* Wr = (const float*)d_in[6];  const float* br = (const float*)d_in[7];
  const float* Wh = (const float*)d_in[8];  const float* bh = (const float*)d_in[9];
  const float* Wlz = (const float*)d_in[10]; const float* blz = (const float*)d_in[11];
  const float* Wlr = (const float*)d_in[12]; const float* blr = (const float*)d_in[13];
  const float* Wlh = (const float*)d_in[14]; const float* blh = (const float*)d_in[15];
  const float* Whead = (const float*)d_in[16]; const float* bhead = (const float*)d_in[17];

  const int N = in_sizes[0] / 64;
  const int E = in_sizes[1];

  // Workspace carve-out (256B aligned).
  char* ws = (char*)d_ws;
  size_t off = 0;
  auto bump = [&](size_t bytes) {
    size_t o = off;
    off += (bytes + 255) & ~(size_t)255;
    return o;
  };
  float*  dinv = (float*)(ws + bump((size_t)N * 4));
  __bf16* hB   = (__bf16*)(ws + bump((size_t)3 * N * 128 * 2));
  float*  agg  = (float*)(ws + bump((size_t)3 * N * 128 * 4));
  __bf16* fw1  = (__bf16*)(ws + bump((size_t)3 * 64 * 128 * 2));
  __bf16* fwl  = (__bf16*)(ws + bump((size_t)3 * 256 * 128 * 2));
  __bf16* fhd  = (__bf16*)(ws + bump((size_t)128 * 64 * 2));

  const int tb = 256;

  // 1) symmetric-norm degrees
  k_deg_init<<<(N + tb - 1) / tb, tb, 0, stream>>>(dinv, N);
  k_deg_atomic<<<(E + tb - 1) / tb, tb, 0, stream>>>(edge_idx, edge_w, dinv, E);
  k_dinv<<<(N + tb - 1) / tb, tb, 0, stream>>>(dinv, N);

  // 2) swizzle weights into WMMA B-fragment layout
  k_prep<<<(64 * 128 + tb - 1) / tb, tb, 0, stream>>>(Wz, fw1,          64, 128);
  k_prep<<<(64 * 128 + tb - 1) / tb, tb, 0, stream>>>(Wr, fw1 + 8192,   64, 128);
  k_prep<<<(64 * 128 + tb - 1) / tb, tb, 0, stream>>>(Wh, fw1 + 16384,  64, 128);
  k_prep<<<(256 * 128 + tb - 1) / tb, tb, 0, stream>>>(Wlz, fwl,           256, 128);
  k_prep<<<(256 * 128 + tb - 1) / tb, tb, 0, stream>>>(Wlr, fwl + 32768,   256, 128);
  k_prep<<<(256 * 128 + tb - 1) / tb, tb, 0, stream>>>(Wlh, fwl + 65536,   256, 128);
  k_prep<<<(128 * 64 + tb - 1) / tb, tb, 0, stream>>>(Whead, fhd, 128, 64);

  // 3) h_g = X @ W_g   (WMMA, bf16 out)
  const int tiles = (N + 15) / 16;
  k_gemm1<<<(tiles + 3) / 4, 128, 0, stream>>>(node_feat, fw1, hB, N);

  // 4) aggregation: self-loop init + edge atomics
  size_t quads = (size_t)3 * N * 32;
  k_agg_init<<<(int)((quads + tb - 1) / tb), tb, 0, stream>>>(dinv, hB, agg, N);
  long long waves = ((long long)E + EPW - 1) / EPW;
  long long sthreads = waves * 32;
  k_scatter<<<(int)((sthreads + tb - 1) / tb), tb, 0, stream>>>(
      edge_idx, edge_w, dinv, hB, agg, E, N);

  // 5) fused gates + head
  float* outY = (float*)d_out;
  float* outH = outY + (size_t)N * 64;
  k_gates<<<(tiles + 1) / 2, 64, 0, stream>>>(agg, hidden, bz, br, bh, fwl,
                                              blz, blr, blh, fhd, bhead,
                                              outY, outH, N);
}
